// LinearAttention_54984171323640
// MI455X (gfx1250) — matrix operand order repaired and checked by reading, not compiled, not run on hardware
//
#include <hip/hip_runtime.h>
#include <hip/hip_bf16.h>

// ---------------------------------------------------------------------------
// LinearAttention for MI455X (gfx1250, wave32, WMMA bf16 16x16x32)
//   cvt: w_qkv -> bf16 workspace (weights re-used by every WG)
//   p1:  k = Wk@x, per-row running max           -> mp partials
//   r1:  reduce mp -> m (row max of k)
//   p2:  k,v = Wkv@x; e=exp(k-m); ctx += e @ (v/N)^T; Z partials (WMMA)
//   r2:  reduce Z, reduce+normalize ctx, fold with W_out -> M_b (bf16)
//   p3:  q = Wq@x; softmax over d; out = M_b@qs + bias; LayerNorm; NT store
// HBM roofline: x streams once (67MB), stays L2-resident (192MB) for p2/p3;
// out 67MB written non-temporally  => ~134MB ~= 5.8us at 23.3 TB/s.
// ---------------------------------------------------------------------------

typedef __bf16 bf16_t;
typedef __attribute__((ext_vector_type(16))) __bf16 v16bf;
typedef __attribute__((ext_vector_type(8)))  __bf16 bf16x8;
typedef __attribute__((ext_vector_type(8)))  float  v8f;

#define BATCH   8
#define CDIM    128
#define NTOT    16384
#define NSPLIT  64      // column splits per batch
#define WGCOLS  256     // columns per workgroup
#define SUBCOLS 64      // columns per inner sub-tile
#define NSUB    4
#define XLDA    136     // padded LDS stride for xT/qsT (272B rows: 16B-aligned, bank-spread)
#define ELDA    72      // padded LDS stride for eL/vL
#define QLDA    68      // padded LDS stride for qof (f32)

#if defined(__has_builtin)
#if __has_builtin(__builtin_amdgcn_global_load_async_to_lds_b128) && \
    __has_builtin(__builtin_amdgcn_s_wait_asynccnt)
#define USE_ASYNC_COPY 1
#endif
#endif
#ifndef USE_ASYNC_COPY
#define USE_ASYNC_COPY 0
#endif

#if USE_ASYNC_COPY
typedef int v4i_t __attribute__((__vector_size__(16)));
typedef __attribute__((address_space(1))) v4i_t* as1_v4i;
typedef __attribute__((address_space(3))) v4i_t* as3_v4i;
#endif

__device__ __forceinline__ v8f zero8() {
  v8f z;
#pragma unroll
  for (int i = 0; i < 8; ++i) z[i] = 0.0f;
  return z;
}

// A fragment (16x32 bf16, row-major src [M][lda]): lane m=lane&15, half=lane>>4
// holds K chunks [half*8, half*8+8) and [16+half*8, 16+half*8+8).
__device__ __forceinline__ v16bf frag_a(const bf16_t* base, int lda, int lane) {
  int m = lane & 15, hp = lane >> 4;
  const bf16_t* p = base + m * lda + hp * 8;
  bf16x8 lo = *(const bf16x8*)(p);
  bf16x8 hi = *(const bf16x8*)(p + 16);
  v16bf r;
#pragma unroll
  for (int i = 0; i < 8; ++i) { r[i] = lo[i]; r[i + 8] = hi[i]; }
  return r;
}

// B fragment (32x16 bf16, src stored [N][ldb] with K contiguous): lane
// n=lane&15 holds K chunks [half*8, half*8+8) and [16+half*8, 16+half*8+8).
__device__ __forceinline__ v16bf frag_b(const bf16_t* base, int ldb, int lane) {
  int n = lane & 15, hp = lane >> 4;
  const bf16_t* p = base + n * ldb + hp * 8;
  bf16x8 lo = *(const bf16x8*)(p);
  bf16x8 hi = *(const bf16x8*)(p + 16);
  v16bf r;
#pragma unroll
  for (int i = 0; i < 8; ++i) { r[i] = lo[i]; r[i + 8] = hi[i]; }
  return r;
}

__device__ __forceinline__ v8f wmma_bf16(v16bf a, v16bf b, v8f c) {
  return __builtin_amdgcn_wmma_f32_16x16x32_bf16(false, a, false, b,
                                                 (short)0, c, false, false);
}

// Copy pre-converted bf16 weight tile global->LDS (16B granules).
// Async LDS-copy path (ASYNCcnt) when the gfx1250 builtins are available.
__device__ __forceinline__ void copy_w_to_lds(bf16_t* dstL, const bf16_t* srcG,
                                              int nElems) {
#if USE_ASYNC_COPY
  for (int i = threadIdx.x * 8; i < nElems; i += 256 * 8) {
    bf16_t* s = const_cast<bf16_t*>(srcG + i);
    bf16_t* d = dstL + i;
    __builtin_amdgcn_global_load_async_to_lds_b128((as1_v4i)s, (as3_v4i)d, 0, 0);
  }
  __builtin_amdgcn_s_wait_asynccnt(0);
#else
  for (int i = threadIdx.x * 8; i < nElems; i += 256 * 8)
    *(bf16x8*)(dstL + i) = *(const bf16x8*)(srcG + i);
#endif
}

// Load 64-column x tile transposed into LDS as bf16: xT[col][chan], two
// channels packed per 32-bit LDS store, padded stride.
__device__ __forceinline__ void load_xT(bf16_t* xT, const float* xb, int n0) {
  int jj = threadIdx.x & 63, g0 = threadIdx.x >> 6;
  const float* col = xb + n0 + jj;
#pragma unroll
  for (int g = g0; g < 64; g += 4) {
    union { bf16_t h[2]; unsigned u; } p;
    p.h[0] = (bf16_t)col[(size_t)(2 * g) * NTOT];
    p.h[1] = (bf16_t)col[(size_t)(2 * g + 1) * NTOT];
    *(unsigned*)(&xT[jj * XLDA + 2 * g]) = p.u;
  }
}

// ---------------- weight conversion ----------------------------------------
__global__ void k_cvt_w(const float* __restrict__ w_qkv,
                        bf16_t* __restrict__ wbf) {
  int i = blockIdx.x * 256 + threadIdx.x;
  if (i < 384 * 128) wbf[i] = (bf16_t)w_qkv[i];
}

// ---------------- pass 1: row max of k -------------------------------------
__global__ __launch_bounds__(256) void k_rowmax(const float* __restrict__ x,
                                                const bf16_t* __restrict__ wbf,
                                                float* __restrict__ mp) {
  __shared__ __align__(16) bf16_t Wl[128 * 128];
  __shared__ __align__(16) bf16_t xT[SUBCOLS * XLDA];
  int b = blockIdx.x >> 6, split = blockIdx.x & 63;
  copy_w_to_lds(Wl, wbf + 128 * 128, 128 * 128);  // k rows
  int wave = threadIdx.x >> 5, lane = threadIdx.x & 31, hp = lane >> 4;
  const float* xb = x + (size_t)b * CDIM * NTOT;
  float best[8];
#pragma unroll
  for (int r = 0; r < 8; ++r) best[r] = -3.402823466e38f;

  for (int sub = 0; sub < NSUB; ++sub) {
    int n0 = split * WGCOLS + sub * SUBCOLS;
    __syncthreads();
    load_xT(xT, xb, n0);
    if (sub + 1 < NSUB)
      __builtin_prefetch(&xb[(size_t)(threadIdx.x >> 6) * NTOT + n0 + SUBCOLS +
                             (threadIdx.x & 63)], 0, 1);
    __syncthreads();
    v8f acc[4];
#pragma unroll
    for (int nt = 0; nt < 4; ++nt) acc[nt] = zero8();
#pragma unroll
    for (int ks = 0; ks < 4; ++ks) {
      v16bf af = frag_a(Wl + (wave * 16) * 128 + ks * 32, 128, lane);
#pragma unroll
      for (int nt = 0; nt < 4; ++nt) {
        v16bf bfv = frag_b(xT + (nt * 16) * XLDA + ks * 32, XLDA, lane);
        acc[nt] = wmma_bf16(af, bfv, acc[nt]);
      }
    }
#pragma unroll
    for (int nt = 0; nt < 4; ++nt)
#pragma unroll
      for (int r = 0; r < 8; ++r) best[r] = fmaxf(best[r], acc[nt][r]);
  }
#pragma unroll
  for (int r = 0; r < 8; ++r) {
    float v = best[r];
#pragma unroll
    for (int mdl = 1; mdl < 16; mdl <<= 1) v = fmaxf(v, __shfl_xor(v, mdl, 32));
    if ((lane & 15) == 0)
      mp[((size_t)(b * NSPLIT + split)) * 128 + wave * 16 + hp * 8 + r] = v;
  }
}

__global__ void k_reduce_max(const float* __restrict__ mp, float* __restrict__ m) {
  int b = blockIdx.x, row = threadIdx.x;
  float v = -3.402823466e38f;
  for (int s = 0; s < NSPLIT; ++s)
    v = fmaxf(v, mp[((size_t)(b * NSPLIT + s)) * 128 + row]);
  m[b * 128 + row] = v;
}

// ---------------- pass 2: context + Z --------------------------------------
__global__ __launch_bounds__(256) void k_context(const float* __restrict__ x,
                                                 const bf16_t* __restrict__ wbf,
                                                 const float* __restrict__ mrow,
                                                 float* __restrict__ ctxp,
                                                 float* __restrict__ Zp) {
  __shared__ __align__(16) bf16_t Wl[256 * 128];   // k rows 0..127, v rows 128..255
  __shared__ __align__(16) bf16_t xT[SUBCOLS * XLDA];
  __shared__ __align__(16) bf16_t eL[128 * ELDA];
  __shared__ __align__(16) bf16_t vL[128 * ELDA];
  __shared__ float mloc[128];
  int b = blockIdx.x >> 6, split = blockIdx.x & 63;
  copy_w_to_lds(Wl, wbf + 128 * 128, 256 * 128);   // k+v rows
  if (threadIdx.x < 128) mloc[threadIdx.x] = mrow[b * 128 + threadIdx.x];
  int wave = threadIdx.x >> 5, lane = threadIdx.x & 31, hp = lane >> 4;
  const float* xb = x + (size_t)b * CDIM * NTOT;
  const float invN = 1.0f / (float)NTOT;

  float zsum[8];
#pragma unroll
  for (int r = 0; r < 8; ++r) zsum[r] = 0.0f;
  v8f cacc[2];
  cacc[0] = zero8();
  cacc[1] = zero8();

  for (int sub = 0; sub < NSUB; ++sub) {
    int n0 = split * WGCOLS + sub * SUBCOLS;
    __syncthreads();
    load_xT(xT, xb, n0);
    __syncthreads();

    // k strip for this wave: global k rows wave*16 .. wave*16+15
    v8f acc[4];
#pragma unroll
    for (int nt = 0; nt < 4; ++nt) acc[nt] = zero8();
#pragma unroll
    for (int ks = 0; ks < 4; ++ks) {
      v16bf af = frag_a(Wl + (wave * 16) * 128 + ks * 32, 128, lane);
#pragma unroll
      for (int nt = 0; nt < 4; ++nt) {
        v16bf bfv = frag_b(xT + (nt * 16) * XLDA + ks * 32, XLDA, lane);
        acc[nt] = wmma_bf16(af, bfv, acc[nt]);
      }
    }
#pragma unroll
    for (int nt = 0; nt < 4; ++nt)
#pragma unroll
      for (int r = 0; r < 8; ++r) {
        int row = wave * 16 + hp * 8 + r;
        float e = __expf(acc[nt][r] - mloc[row]);
        zsum[r] += e;
        eL[row * ELDA + nt * 16 + (lane & 15)] = (bf16_t)e;
      }

    // v strip (Wl rows 128+wave*16)
#pragma unroll
    for (int nt = 0; nt < 4; ++nt) acc[nt] = zero8();
#pragma unroll
    for (int ks = 0; ks < 4; ++ks) {
      v16bf af = frag_a(Wl + (128 + wave * 16) * 128 + ks * 32, 128, lane);
#pragma unroll
      for (int nt = 0; nt < 4; ++nt) {
        v16bf bfv = frag_b(xT + (nt * 16) * XLDA + ks * 32, XLDA, lane);
        acc[nt] = wmma_bf16(af, bfv, acc[nt]);
      }
    }
#pragma unroll
    for (int nt = 0; nt < 4; ++nt)
#pragma unroll
      for (int r = 0; r < 8; ++r) {
        int row = wave * 16 + hp * 8 + r;
        vL[row * ELDA + nt * 16 + (lane & 15)] = (bf16_t)(acc[nt][r] * invN);
      }
    __syncthreads();

    // ctx[head][dk][dv] += e_h @ v_h^T ; 16 tile-jobs over 8 waves (2 each)
#pragma unroll
    for (int t = 0; t < 2; ++t) {
      int jjb = wave * 2 + t;
      int head = jjb >> 2, tm = (jjb >> 1) & 1, tn = jjb & 1;
#pragma unroll
      for (int ks = 0; ks < 2; ++ks) {
        v16bf af = frag_a(eL + (head * 32 + tm * 16) * ELDA + ks * 32, ELDA, lane);
        v16bf bfv = frag_b(vL + (head * 32 + tn * 16) * ELDA + ks * 32, ELDA, lane);
        cacc[t] = wmma_bf16(af, bfv, cacc[t]);
      }
    }
  }

  size_t base = ((size_t)(b * NSPLIT + split)) * 4096;
#pragma unroll
  for (int t = 0; t < 2; ++t) {
    int jjb = wave * 2 + t;
    int head = jjb >> 2, tm = (jjb >> 1) & 1, tn = jjb & 1;
#pragma unroll
    for (int r = 0; r < 8; ++r)
      ctxp[base + head * 1024 + (tm * 16 + hp * 8 + r) * 32 + tn * 16 + (lane & 15)] =
          cacc[t][r];
  }
#pragma unroll
  for (int r = 0; r < 8; ++r) {
    float v = zsum[r];
#pragma unroll
    for (int mdl = 1; mdl < 16; mdl <<= 1) v += __shfl_xor(v, mdl, 32);
    if ((lane & 15) == 0)
      Zp[((size_t)(b * NSPLIT + split)) * 128 + wave * 16 + hp * 8 + r] = v;
  }
}

__global__ void k_reduce_sum(const float* __restrict__ Zp, float* __restrict__ Z) {
  int b = blockIdx.x, row = threadIdx.x;
  float v = 0.0f;
  for (int s = 0; s < NSPLIT; ++s)
    v += Zp[((size_t)(b * NSPLIT + s)) * 128 + row];
  Z[b * 128 + row] = v;
}

__global__ void k_reduce_ctx(const float* __restrict__ ctxp,
                             const float* __restrict__ Z,
                             float* __restrict__ ctxf) {
  int b = blockIdx.x;
  for (int e = threadIdx.x; e < 4096; e += 256) {
    float s = 0.0f;
    for (int sp = 0; sp < NSPLIT; ++sp)
      s += ctxp[((size_t)(b * NSPLIT + sp)) * 4096 + e];
    int head = e >> 10, dk = (e >> 5) & 31;
    ctxf[(size_t)b * 4096 + e] = s / Z[b * 128 + head * 32 + dk];
  }
}

// M_b[c][h*32+dk] = sum_dv w_out[c][h*32+dv] * ctx[b][h][dk][dv]  (bf16 out)
__global__ void k_fuse_m(const float* __restrict__ ctxf,
                         const float* __restrict__ w_out,
                         bf16_t* __restrict__ Mfb) {
  int b = blockIdx.x;
  for (int e = threadIdx.x; e < 16384; e += 256) {
    int c = e >> 7, dkg = e & 127, h = dkg >> 5, dk = dkg & 31;
    float s = 0.0f;
#pragma unroll 8
    for (int dv = 0; dv < 32; ++dv)
      s += w_out[c * 128 + h * 32 + dv] * ctxf[(size_t)b * 4096 + h * 1024 + dk * 32 + dv];
    Mfb[(size_t)b * 16384 + e] = (bf16_t)s;
  }
}

// ---------------- pass 3: q softmax, fused out-proj, LayerNorm -------------
__global__ __launch_bounds__(256) void k_output(const float* __restrict__ x,
                                                const bf16_t* __restrict__ wbf,
                                                const bf16_t* __restrict__ Mfb,
                                                const float* __restrict__ b_out,
                                                const float* __restrict__ alpha,
                                                const float* __restrict__ beta,
                                                float* __restrict__ out) {
  __shared__ __align__(16) bf16_t Wl[128 * 128];        // q weights
  __shared__ __align__(16) bf16_t Ml[128 * 128];        // fused ctx*W_out
  __shared__ __align__(16) bf16_t xT[SUBCOLS * XLDA];
  __shared__ __align__(16) bf16_t qsT[SUBCOLS * XLDA];  // softmaxed q, [col][dkg]
  __shared__ float qof[128 * QLDA];                     // q pre-softmax / out pre-LN
  __shared__ float biasL[128], alphaL[128], betaL[128];
  __shared__ float meanL[SUBCOLS], rstdL[SUBCOLS];
  int b = blockIdx.x >> 6, split = blockIdx.x & 63;
  copy_w_to_lds(Wl, wbf, 128 * 128);                    // q rows
  copy_w_to_lds(Ml, Mfb + (size_t)b * 16384, 128 * 128);
  if (threadIdx.x < 128) {
    biasL[threadIdx.x] = b_out[threadIdx.x];
    alphaL[threadIdx.x] = alpha[threadIdx.x];
    betaL[threadIdx.x] = beta[threadIdx.x];
  }
  int wave = threadIdx.x >> 5, lane = threadIdx.x & 31, hp = lane >> 4;
  const float* xb = x + (size_t)b * CDIM * NTOT;
  float* ob = out + (size_t)b * CDIM * NTOT;
  const float scale = 0.1767766952966369f;  // 32^-0.5

  for (int sub = 0; sub < NSUB; ++sub) {
    int n0 = split * WGCOLS + sub * SUBCOLS;
    __syncthreads();
    load_xT(xT, xb, n0);
    __syncthreads();

    // GEMM1: q tile
    v8f acc[4];
#pragma unroll
    for (int nt = 0; nt < 4; ++nt) acc[nt] = zero8();
#pragma unroll
    for (int ks = 0; ks < 4; ++ks) {
      v16bf af = frag_a(Wl + (wave * 16) * 128 + ks * 32, 128, lane);
#pragma unroll
      for (int nt = 0; nt < 4; ++nt) {
        v16bf bfv = frag_b(xT + (nt * 16) * XLDA + ks * 32, XLDA, lane);
        acc[nt] = wmma_bf16(af, bfv, acc[nt]);
      }
    }
#pragma unroll
    for (int nt = 0; nt < 4; ++nt)
#pragma unroll
      for (int r = 0; r < 8; ++r)
        qof[(wave * 16 + hp * 8 + r) * QLDA + nt * 16 + (lane & 15)] = acc[nt][r];
    __syncthreads();

    // softmax over d=32 per (head, col); write bf16 qsT[col][h*32+i]
    {
      int col = threadIdx.x & 63, head = threadIdx.x >> 6;
      float tmp[32];
      float mx = -3.402823466e38f;
#pragma unroll
      for (int i = 0; i < 32; ++i) {
        tmp[i] = qof[(head * 32 + i) * QLDA + col];
        mx = fmaxf(mx, tmp[i]);
      }
      float s = 0.0f;
#pragma unroll
      for (int i = 0; i < 32; ++i) { tmp[i] = __expf(tmp[i] - mx); s += tmp[i]; }
      float inv = scale / s;
#pragma unroll
      for (int i = 0; i < 32; ++i)
        qsT[col * XLDA + head * 32 + i] = (bf16_t)(tmp[i] * inv);
    }
    __syncthreads();

    // GEMM2: out = M_b @ qs  (+bias into LDS for LN)
    v8f acc2[4];
#pragma unroll
    for (int nt = 0; nt < 4; ++nt) acc2[nt] = zero8();
#pragma unroll
    for (int ks = 0; ks < 4; ++ks) {
      v16bf af = frag_a(Ml + (wave * 16) * 128 + ks * 32, 128, lane);
#pragma unroll
      for (int nt = 0; nt < 4; ++nt) {
        v16bf bfv = frag_b(qsT + (nt * 16) * XLDA + ks * 32, XLDA, lane);
        acc2[nt] = wmma_bf16(af, bfv, acc2[nt]);
      }
    }
#pragma unroll
    for (int nt = 0; nt < 4; ++nt)
#pragma unroll
      for (int r = 0; r < 8; ++r) {
        int row = wave * 16 + hp * 8 + r;
        qof[row * QLDA + nt * 16 + (lane & 15)] = acc2[nt][r] + biasL[row];
      }
    __syncthreads();

    // LayerNorm stats per column
    if (threadIdx.x < SUBCOLS) {
      int col = threadIdx.x;
      float s = 0.0f, s2 = 0.0f;
      for (int c = 0; c < 128; ++c) {
        float v = qof[c * QLDA + col];
        s += v;
        s2 += v * v;
      }
      float mean = s * (1.0f / 128.0f);
      float var = s2 * (1.0f / 128.0f) - mean * mean;
      meanL[col] = mean;
      rstdL[col] = rsqrtf(var + 1e-5f);
    }
    __syncthreads();

    // normalize + affine + coalesced non-temporal store (protect x in L2)
    {
      int col = threadIdx.x & 63, cc0 = threadIdx.x >> 6;
      for (int c = cc0; c < 128; c += 4) {
        float v = (qof[c * QLDA + col] - meanL[col]) * rstdL[col] * alphaL[c] + betaL[c];
        __builtin_nontemporal_store(v, &ob[(size_t)c * NTOT + n0 + col]);
      }
    }
  }
}

// ---------------------------------------------------------------------------
extern "C" void kernel_launch(void* const* d_in, const int* in_sizes, int n_in,
                              void* d_out, int out_size, void* d_ws, size_t ws_size,
                              hipStream_t stream) {
  (void)in_sizes; (void)n_in; (void)out_size; (void)ws_size;
  const float* x     = (const float*)d_in[0];
  const float* w_qkv = (const float*)d_in[1];
  const float* w_out = (const float*)d_in[2];
  const float* b_out = (const float*)d_in[3];
  const float* alpha = (const float*)d_in[4];
  const float* beta  = (const float*)d_in[5];
  float* out = (float*)d_out;

  float* ws   = (float*)d_ws;
  float* mp   = ws;                 // 8*64*128      = 65536 f
  float* m    = mp + 65536;         // 8*128         = 1024 f
  float* Zp   = m + 1024;           // 8*64*128      = 65536 f
  float* Z    = Zp + 65536;         // 8*128         = 1024 f
  float* ctxp = Z + 1024;           // 8*64*4096     = 2097152 f
  float* ctxf = ctxp + 2097152;     // 8*4096        = 32768 f
  bf16_t* wbf = (bf16_t*)(ctxf + 32768);        // 384*128 bf16
  bf16_t* Mfb = (bf16_t*)(ctxf + 32768 + 24576);// 8*128*128 bf16

  dim3 blk(256);
  k_cvt_w   <<<dim3(192), blk, 0, stream>>>(w_qkv, wbf);
  k_rowmax  <<<dim3(BATCH * NSPLIT), blk, 0, stream>>>(x, wbf, mp);
  k_reduce_max<<<dim3(BATCH), dim3(128), 0, stream>>>(mp, m);
  k_context <<<dim3(BATCH * NSPLIT), blk, 0, stream>>>(x, wbf, m, ctxp, Zp);
  k_reduce_sum<<<dim3(BATCH), dim3(128), 0, stream>>>(Zp, Z);
  k_reduce_ctx<<<dim3(BATCH), blk, 0, stream>>>(ctxp, Z, ctxf);
  k_fuse_m  <<<dim3(BATCH), blk, 0, stream>>>(ctxf, w_out, Mfb);
  k_output  <<<dim3(BATCH * NSPLIT), blk, 0, stream>>>(x, wbf, Mfb, b_out, alpha,
                                                       beta, out);
}
